// Base_1348619731207
// MI455X (gfx1250) — compile-verified, matrix operand order checked by reading, not run on hardware
//
#include <hip/hip_runtime.h>

typedef __attribute__((ext_vector_type(16))) __bf16 v16bf;
typedef __attribute__((ext_vector_type(8)))  float  v8f;

#define Bn   128
#define Nn   512
#define Hn   256
#define DSn  128
#define D1n  256
#define D2n  128
#define Gn   32
#define OUTW (Gn + Nn)   // 544

// ---------------------------------------------------------------------------
// Kernel 1: graph head. One block per graph. Tiny compute; bandwidth-bound on
// re-reading x. Plain VALU dot products (WMMA would not help at this size).
// ---------------------------------------------------------------------------
__global__ __launch_bounds__(256, 1)
void graph_head_kernel(const float* __restrict__ x,
                       const float* __restrict__ gs_w1, const float* __restrict__ gs_b1,
                       const float* __restrict__ gs_w2, const float* __restrict__ gs_b2,
                       const float* __restrict__ gh_w1, const float* __restrict__ gh_b1,
                       const float* __restrict__ gh_w2, const float* __restrict__ gh_b2,
                       const float* __restrict__ gh_w3, const float* __restrict__ gh_b3,
                       float* __restrict__ out)
{
    __shared__ float s1[256];
    __shared__ float s2[256];
    const int g = blockIdx.x;
    const int t = threadIdx.x;

    // mean over N rows of column t, then ReLU (reference applies relu before gs_w1)
    const float* xg = x + (size_t)g * Nn * Hn + t;
    float sum = 0.f;
    #pragma unroll 8
    for (int r = 0; r < Nn; ++r) sum += xg[(size_t)r * Hn];
    s1[t] = fmaxf(sum * (1.0f / (float)Nn), 0.f);
    __syncthreads();

    // gs_w1 [256,128] + b (no relu between gs_w1 and gs_w2)
    float v = 0.f;
    if (t < DSn) {
        v = gs_b1[t];
        for (int k = 0; k < Hn; ++k) v += s1[k] * gs_w1[k * DSn + t];
    }
    __syncthreads();
    if (t < DSn) s2[t] = v;
    __syncthreads();

    // gs_w2 [128,128] + b, relu
    if (t < DSn) {
        v = gs_b2[t];
        for (int k = 0; k < DSn; ++k) v += s2[k] * gs_w2[k * DSn + t];
        v = fmaxf(v, 0.f);
    }
    __syncthreads();
    if (t < DSn) s1[t] = v;
    __syncthreads();

    // gh_w1 [128,256] + b, relu
    {
        float u = gh_b1[t];
        for (int k = 0; k < DSn; ++k) u += s1[k] * gh_w1[k * D1n + t];
        u = fmaxf(u, 0.f);
        __syncthreads();
        s2[t] = u;
        __syncthreads();
    }

    // gh_w2 [256,128] + b, relu
    if (t < D2n) {
        v = gh_b2[t];
        for (int k = 0; k < D1n; ++k) v += s2[k] * gh_w2[k * D2n + t];
        v = fmaxf(v, 0.f);
    }
    __syncthreads();
    if (t < D2n) s1[t] = v;
    __syncthreads();

    // gh_w3 [128,32] + b
    if (t < Gn) {
        float o = gh_b3[t];
        for (int k = 0; k < D2n; ++k) o += s1[k] * gh_w3[k * Gn + t];
        out[(size_t)g * OUTW + t] = o;
    }
}

// ---------------------------------------------------------------------------
// Kernel 2: per-node MLP heads. One block per node n (512 blocks, 8 waves).
// bf16 WMMA (v_wmma_f32_16x16x32_bf16), fp32->bf16 conversion during LDS
// staging, weight chunks staged transposed so B-fragments are contiguous
// 32-byte LDS loads per lane. Wave w owns batch rows [16w, 16w+16).
// ---------------------------------------------------------------------------
__global__ __launch_bounds__(256, 1)
void node_heads_kernel(const float* __restrict__ x,
                       const float* __restrict__ w1, const float* __restrict__ b1,
                       const float* __restrict__ w2, const float* __restrict__ b2,
                       const float* __restrict__ w3, const float* __restrict__ b3,
                       float* __restrict__ out)
{
    extern __shared__ char smem[];
    __bf16* sH1 = (__bf16*)smem;                  // [128][256] h1 (reused for h2)
    __bf16* sX  = (__bf16*)(smem + 65536);        // [128][32]  X chunk
    __bf16* sWt = (__bf16*)(smem + 73728);        // [256][32]  W chunk, transposed (N-major)
    float*  sB1 = (float*)(smem + 90112);         // [256]
    float*  sB2 = (float*)(smem + 91136);         // [128]
    float*  sW3 = (float*)(smem + 91648);         // [128]

    const int n    = blockIdx.x;
    const int tid  = threadIdx.x;
    const int lane = tid & 31;
    const int wav  = tid >> 5;          // 0..7 -> M tile
    const int half = lane >> 4;         // lane group (K/M split per ISA layouts)
    const int l    = lane & 15;
    const int mbase = wav * 16;

    const float* W1 = w1 + (size_t)n * Hn * D1n;   // [256 K][256 N]
    const float* W2 = w2 + (size_t)n * D1n * D2n;  // [256 K][128 N]

    // stage biases / w3 (visible after first barrier of stage-1 loop)
    sB1[tid] = b1[(size_t)n * D1n + tid];
    if (tid < D2n) sB2[tid] = b2[(size_t)n * D2n + tid];
    if (tid < D2n) sW3[tid] = w3[(size_t)n * D2n + tid];

    v8f zero = {};
    v8f acc[16];
    #pragma unroll
    for (int i = 0; i < 16; ++i) acc[i] = zero;

    // ================= stage 1: h1 = relu(X_n @ W1_n + b1)  (M=128,K=256,N=256)
    for (int kc = 0; kc < Hn / 32; ++kc) {
        __syncthreads();                       // previous chunk's compute done
        // stage X chunk [128 b][32 k] as bf16 (coalesced over k)
        #pragma unroll
        for (int it = 0; it < 16; ++it) {
            int idx = tid + it * 256;
            int b = idx >> 5, hh = idx & 31;
            sX[idx] = (__bf16)x[((size_t)b * Nn + n) * Hn + kc * 32 + hh];
        }
        // stage W1 chunk transposed: sWt[d][kk]  (coalesced global read over d)
        #pragma unroll
        for (int it = 0; it < 32; ++it) {
            int idx = tid + it * 256;
            int kk = idx >> 8, d = idx & 255;
            sWt[d * 32 + kk] = (__bf16)W1[(size_t)(kc * 32 + kk) * D1n + d];
        }
        if (kc + 1 < Hn / 32)
            __builtin_prefetch(W1 + (size_t)(kc + 1) * 32 * D1n + tid * 32, 0, 1);
        __syncthreads();

        // A fragment: lane<16 -> M=mbase+l,K=0..15 ; lane>=16 -> K=16..31
        v16bf a = *(const v16bf*)(sX + (mbase + l) * 32 + half * 16);
        #pragma unroll
        for (int nt = 0; nt < 16; ++nt) {
            // B fragment from transposed buffer: lane<16 -> N=nt*16+l,K=0..15
            v16bf bb = *(const v16bf*)(sWt + (nt * 16 + l) * 32 + half * 16);
            acc[nt] = __builtin_amdgcn_wmma_f32_16x16x32_bf16(
                false, a, false, bb, (short)0, acc[nt], false, false);
        }
    }

    // epilogue: relu + bias, store h1 as bf16 (C/D layout -> row-major LDS)
    #pragma unroll
    for (int nt = 0; nt < 16; ++nt) {
        int col = nt * 16 + l;
        float bias = sB1[col];
        #pragma unroll
        for (int r = 0; r < 8; ++r) {
            int row = mbase + half * 8 + r;
            float v = acc[nt][r] + bias;
            sH1[row * 256 + col] = (__bf16)fmaxf(v, 0.0f);
        }
    }

    // ================= stage 2: h2 = relu(h1 @ W2_n + b2)  (M=128,K=256,N=128)
    v8f acc2[8];
    #pragma unroll
    for (int i = 0; i < 8; ++i) acc2[i] = zero;

    for (int kc = 0; kc < D1n / 32; ++kc) {
        __syncthreads();                       // h1 stores + prior reads of sWt done
        #pragma unroll
        for (int it = 0; it < 16; ++it) {
            int idx = tid + it * 256;          // 32*128 elements
            int kk = idx >> 7, d = idx & 127;
            sWt[d * 32 + kk] = (__bf16)W2[(size_t)(kc * 32 + kk) * D2n + d];
        }
        if (kc + 1 < D1n / 32)
            __builtin_prefetch(W2 + (size_t)(kc + 1) * 32 * D2n + tid * 16, 0, 1);
        __syncthreads();

        // A fragment straight out of row-major bf16 h1
        v16bf a = *(const v16bf*)(sH1 + (mbase + l) * 256 + kc * 32 + half * 16);
        #pragma unroll
        for (int nt = 0; nt < 8; ++nt) {
            v16bf bb = *(const v16bf*)(sWt + (nt * 16 + l) * 32 + half * 16);
            acc2[nt] = __builtin_amdgcn_wmma_f32_16x16x32_bf16(
                false, a, false, bb, (short)0, acc2[nt], false, false);
        }
    }

    __syncthreads();                           // everyone done reading sH1
    __bf16* sH2 = sH1;                         // reuse region, stride 128
    #pragma unroll
    for (int nt = 0; nt < 8; ++nt) {
        int col = nt * 16 + l;
        float bias = sB2[col];
        #pragma unroll
        for (int r = 0; r < 8; ++r) {
            int row = mbase + half * 8 + r;
            float v = acc2[nt][r] + bias;
            sH2[row * 128 + col] = (__bf16)fmaxf(v, 0.0f);
        }
    }
    __syncthreads();

    // ================= stage 3: node_out[b] = h2[b,:] . w3 + b3   (GEMV, K=128)
    if (tid < Bn) {
        float sum = b3[n];
        #pragma unroll 8
        for (int k = 0; k < D2n; ++k)
            sum += (float)sH2[tid * 128 + k] * sW3[k];
        out[(size_t)tid * OUTW + Gn + n] = sum;
    }
}

// ---------------------------------------------------------------------------
extern "C" void kernel_launch(void* const* d_in, const int* in_sizes, int n_in,
                              void* d_out, int out_size, void* d_ws, size_t ws_size,
                              hipStream_t stream) {
    (void)in_sizes; (void)n_in; (void)d_ws; (void)ws_size; (void)out_size;

    const float* x     = (const float*)d_in[0];
    // d_in[1] = batch ids (layout is implicit: row = b*N + n) - unused
    const float* gs_w1 = (const float*)d_in[2];
    const float* gs_b1 = (const float*)d_in[3];
    const float* gs_w2 = (const float*)d_in[4];
    const float* gs_b2 = (const float*)d_in[5];
    const float* gh_w1 = (const float*)d_in[6];
    const float* gh_b1 = (const float*)d_in[7];
    const float* gh_w2 = (const float*)d_in[8];
    const float* gh_b2 = (const float*)d_in[9];
    const float* gh_w3 = (const float*)d_in[10];
    const float* gh_b3 = (const float*)d_in[11];
    const float* nh_w1 = (const float*)d_in[12];
    const float* nh_b1 = (const float*)d_in[13];
    const float* nh_w2 = (const float*)d_in[14];
    const float* nh_b2 = (const float*)d_in[15];
    const float* nh_w3 = (const float*)d_in[16];
    const float* nh_b3 = (const float*)d_in[17];
    float* out = (float*)d_out;

    graph_head_kernel<<<Bn, 256, 0, stream>>>(
        x, gs_w1, gs_b1, gs_w2, gs_b2,
        gh_w1, gh_b1, gh_w2, gh_b2, gh_w3, gh_b3, out);

    const size_t lds_bytes = 92160;   // 90 KB dynamic LDS (WGP has 320 KB)
    node_heads_kernel<<<Nn, 256, lds_bytes, stream>>>(
        x, nh_w1, nh_b1, nh_w2, nh_b2, nh_w3, nh_b3, out);
}
